// EfficientGraphAttentionBlock_14491219657124
// MI455X (gfx1250) — matrix operand order repaired and checked by reading, not compiled, not run on hardware
//
#include <hip/hip_runtime.h>

// ---------------------------------------------------------------------------
// EfficientGraphAttentionBlock for MI455X (gfx1250, wave32, WMMA f16->f32)
//
//   wt_convert_kernel  : fp32 weights -> f16, transposed to N-major (11x)
//   node_ln_kernel     : nh = LN(node_features) -> f16 gather table in ws
//   edge_block_kernel  : one workgroup per node; entire edge pipeline fused
//                        in ~219KB dynamic LDS; all GEMMs via
//                        v_wmma_f32_16x16x32_f16
//   node_ffn_kernel    : 16 nodes per block; LN + FFN + residual via WMMA
//
// Round 3: each wave owns an N-tile and computes both M-halves from a single
// B fragment (gemm_block2) -> halves weight traffic per WMMA.
// ---------------------------------------------------------------------------

#define NNODE 8000
#define KNBR  32
#define HDIM  256
#define HEADS 8
#define DHEAD 32
#define DEDIM 128
#define H2    512
#define H3    768
#define BIG_NEG (-1e9f)

typedef _Float16 v16h __attribute__((ext_vector_type(16)));
typedef _Float16 h8v  __attribute__((ext_vector_type(8)));
typedef float    v8f  __attribute__((ext_vector_type(8)));

// ---- fast branch-free transcendentals (CDNA5 TRANS32 units) ---------------
__device__ __forceinline__ float exp_fast(float x) {
#if __has_builtin(__builtin_amdgcn_exp2f)
    return __builtin_amdgcn_exp2f(x * 1.4426950408889634f);   // v_exp_f32
#else
    return __expf(x);
#endif
}

__device__ __forceinline__ float rcp_fast(float x) {
#if __has_builtin(__builtin_amdgcn_rcpf)
    return __builtin_amdgcn_rcpf(x);                          // v_rcp_f32
#else
    return 1.0f / x;
#endif
}

__device__ __forceinline__ float tanh_fast(float x) {
#if __has_builtin(__builtin_amdgcn_tanhf)
    return __builtin_amdgcn_tanhf(x);                         // v_tanh_f32 (gfx1250)
#elif __has_builtin(__builtin_amdgcn_tanh_f32)
    return __builtin_amdgcn_tanh_f32(x);
#else
    float xc = fminf(fmaxf(x, -15.0f), 15.0f);
    float e  = exp_fast(2.0f * xc);
    return 1.0f - 2.0f * rcp_fast(e + 1.0f);
#endif
}

__device__ __forceinline__ float sigmoid_f(float x) {
    return rcp_fast(1.0f + exp_fast(-x));
}

__device__ __forceinline__ float gelu_f(float x) {
    const float c0 = 0.7978845608028654f; // sqrt(2/pi)
    return 0.5f * x * (1.0f + tanh_fast(c0 * (x + 0.044715f * x * x * x)));
}

// A fragment: 16x32 f16 (M x K). lane%16 = M row; each lane
// holds K = 16*(lane/16)+0..15 in elems 0..15 (matches B fragment order).
__device__ __forceinline__ v16h load_a_frag(const _Float16* __restrict__ A,
                                            int stride, int m0, int k0, int lane) {
    int m = m0 + (lane & 15);
    const _Float16* p = A + (size_t)m * stride + k0 + ((lane >> 4) << 4);
    h8v lo = *(const h8v*)p;
    h8v hi = *(const h8v*)(p + 8);
    v16h a;
#pragma unroll
    for (int i = 0; i < 8; i++) { a[i] = lo[i]; a[i + 8] = hi[i]; }
    return a;
}

// B fragment: 32x16 f16 (K x N) from an N-major (transposed) matrix:
// row n holds contiguous K. lane%16 = N; elems 0..15 = K = 16*(lane/16)+0..15.
__device__ __forceinline__ v16h load_b_frag(const _Float16* __restrict__ Wt,
                                            int stride, int n0, int k0, int lane) {
    int n = n0 + (lane & 15);
    const _Float16* p = Wt + (size_t)n * stride + k0 + ((lane >> 4) << 4);
    h8v lo = *(const h8v*)p;
    h8v hi = *(const h8v*)(p + 8);
    v16h b;
#pragma unroll
    for (int i = 0; i < 8; i++) { b[i] = lo[i]; b[i + 8] = hi[i]; }
    return b;
}

// B fragment from a K-major matrix (row = K index, e.g. V in attention).
__device__ __forceinline__ v16h load_b_frag_kmajor(const _Float16* __restrict__ B,
                                                   int stride, int n0, int k0, int lane) {
    int n  = n0 + (lane & 15);
    int kb = k0 + ((lane >> 4) << 4);
    v16h b;
#pragma unroll
    for (int i = 0; i < 16; i++) b[i] = B[(size_t)(kb + i) * stride + n];
    return b;
}

__device__ __forceinline__ v8f wmma_acc(v8f c, v16h a, v16h b) {
    return __builtin_amdgcn_wmma_f32_16x16x32_f16(false, a, false, b,
                                                  (short)0, c, false, false);
}

// Dual-M GEMM: one B fragment feeds both 16-row halves of the 32-row A tile.
// c0 accumulates rows 0..15, c1 rows 16..31, for output columns [n0, n0+16).
__device__ __forceinline__ void gemm_block2(v8f& c0, v8f& c1,
                                            const _Float16* __restrict__ A, int strideA,
                                            const _Float16* __restrict__ Wt, int strideW,
                                            int n0, int kw0, int ksteps, int lane) {
#pragma unroll
    for (int ks = 0; ks < ksteps; ks++) {
        v16h b  = load_b_frag(Wt, strideW, n0, kw0 + ks * 32, lane);
        v16h a0 = load_a_frag(A, strideA, 0,  ks * 32, lane);
        v16h a1 = load_a_frag(A, strideA, 16, ks * 32, lane);
        c0 = wmma_acc(c0, a0, b);
        c1 = wmma_acc(c1, a1, b);
    }
}

// Single-M (16-row) GEMM, used by the node FFN.
__device__ __forceinline__ v8f gemm_block(v8f c,
                                          const _Float16* __restrict__ A, int strideA,
                                          const _Float16* __restrict__ Wt, int strideW,
                                          int m0, int n0, int kw0, int ksteps, int lane) {
#pragma unroll
    for (int ks = 0; ks < ksteps; ks++) {
        v16h a = load_a_frag(A, strideA, m0, ks * 32, lane);
        v16h b = load_b_frag(Wt, strideW, n0, kw0 + ks * 32, lane);
        c = wmma_acc(c, a, b);
    }
    return c;
}

// ---------------------------------------------------------------------------
// fp32 [Kd][Nd] -> f16 transposed [Nd][Kd]
__global__ void wt_convert_kernel(const float* __restrict__ W,
                                  _Float16* __restrict__ Wt, int Kd, int Nd) {
    int i = blockIdx.x * blockDim.x + threadIdx.x;
    if (i >= Kd * Nd) return;
    int k = i / Nd, n = i - k * Nd;
    Wt[(size_t)n * Kd + k] = (_Float16)W[i];
}

// ---------------------------------------------------------------------------
// nh = LN(node_features) -> f16 [N][H]
__global__ __launch_bounds__(256)
void node_ln_kernel(const float* __restrict__ nf, const float* __restrict__ g,
                    const float* __restrict__ b, _Float16* __restrict__ nh16) {
    __shared__ float s0[256], s1[256];
    int n = blockIdx.x, t = threadIdx.x;
    float x = nf[(size_t)n * HDIM + t];
    s0[t] = x; s1[t] = x * x;
    __syncthreads();
    for (int off = 128; off > 0; off >>= 1) {
        if (t < off) { s0[t] += s0[t + off]; s1[t] += s1[t + off]; }
        __syncthreads();
    }
    float mu = s0[0] * (1.0f / HDIM);
    float var = s1[0] * (1.0f / HDIM) - mu * mu;
    float rs = rsqrtf(var + 1e-5f);
    nh16[(size_t)n * HDIM + t] = (_Float16)((x - mu) * rs * g[t] + b[t]);
}

// ---------------------------------------------------------------------------
// Fused per-node edge pipeline.
__global__ __launch_bounds__(256)
void edge_block_kernel(
    const float* __restrict__ node_features, const float* __restrict__ edge_features,
    const float* __restrict__ edge_attr_raw, const float* __restrict__ angle,
    const float* __restrict__ ln1e_g, const float* __restrict__ ln1e_b,
    const float* __restrict__ b_ea, const float* __restrict__ b_nh,
    const float* __restrict__ b_eh, const float* __restrict__ b_gate,
    const float* __restrict__ b_cand, const float* __restrict__ attn_scalar,
    const float* __restrict__ b_in, const float* __restrict__ b_out,
    const float* __restrict__ attn_weight,
    const float* __restrict__ ln2e_g, const float* __restrict__ ln2e_b,
    const float* __restrict__ be1, const float* __restrict__ be2,
    const _Float16* __restrict__ Wt_ea, const _Float16* __restrict__ Wt_nh,
    const _Float16* __restrict__ Wt_eh, const _Float16* __restrict__ Wt_gate,
    const _Float16* __restrict__ Wt_cand, const _Float16* __restrict__ Wt_in,
    const _Float16* __restrict__ Wt_out, const _Float16* __restrict__ Wt_e1,
    const _Float16* __restrict__ Wt_e2, const _Float16* __restrict__ nh16,
    const int* __restrict__ neighbor_list, const int* __restrict__ neighbor_counts,
    float* __restrict__ node_resid, float* __restrict__ out_edge) {

    extern __shared__ char smem[];
    // phase-1 region (overlaid by Q/K/V/SC after the message stage)
    _Float16* sEH   = (_Float16*)(smem + 0);          // [32][256]
    _Float16* sNBR  = (_Float16*)(smem + 16 * 1024);  // [32][256]
    _Float16* sEAin = (_Float16*)(smem + 32 * 1024);  // [32][128]
    _Float16* sEA   = (_Float16*)(smem + 40 * 1024);  // [32][256]
    _Float16* sNH   = (_Float16*)(smem + 56 * 1024);  // [32][256]
    _Float16* sEHID = (_Float16*)(smem + 72 * 1024);  // [32][256]  (end 88K)
    _Float16* sQ    = (_Float16*)(smem + 0);          // overlay
    _Float16* sK    = (_Float16*)(smem + 16 * 1024);
    _Float16* sV    = (_Float16*)(smem + 32 * 1024);
    float*    sSC   = (float*)   (smem + 48 * 1024);  // [8][32][32] f32
    _Float16* sMSG  = (_Float16*)(smem + 88 * 1024);  // [32][256]
    _Float16* sP    = (_Float16*)(smem + 104 * 1024); // [8][32][32] f16
    _Float16* sCTX  = (_Float16*)(smem + 120 * 1024); // [32][256]
    float*    sEO   = (float*)   (smem + 136 * 1024); // [32][256] f32
    _Float16* sE2   = (_Float16*)(smem + 168 * 1024); // [32][256]
    _Float16* sT1   = (_Float16*)(smem + 184 * 1024); // [32][512]
    float*    sALPHA= (float*)   (smem + 216 * 1024); // [32][8]
    float*    sRED  = (float*)   (smem + 217 * 1024); // [32][8][2]

    const int n = blockIdx.x;
    const int t = threadIdx.x;
    const int lane = t & 31, wave = t >> 5;
    const int cnt = neighbor_counts[n];

    // ---- Stage 1: LN(edge_features) -> sEH (f16) -------------------------
    {
        int r = t >> 3, q = t & 7;
        const float* src = edge_features + (size_t)n * KNBR * HDIM + r * HDIM + q * 32;
        float xv[32], s0 = 0.f, s1 = 0.f;
#pragma unroll
        for (int j = 0; j < 32; j++) { xv[j] = src[j]; s0 += xv[j]; s1 += xv[j] * xv[j]; }
        sRED[(r * 8 + q) * 2 + 0] = s0; sRED[(r * 8 + q) * 2 + 1] = s1;
        __syncthreads();
        float mu = 0.f, m2 = 0.f;
#pragma unroll
        for (int j = 0; j < 8; j++) { mu += sRED[(r * 8 + j) * 2]; m2 += sRED[(r * 8 + j) * 2 + 1]; }
        mu *= (1.0f / HDIM); m2 = m2 * (1.0f / HDIM) - mu * mu;
        float rs = rsqrtf(m2 + 1e-5f);
#pragma unroll
        for (int j = 0; j < 32; j++) {
            int c = q * 32 + j;
            sEH[r * HDIM + c] = (_Float16)((xv[j] - mu) * rs * ln1e_g[c] + ln1e_b[c]);
        }
    }
    // ---- Stage 2: gather LN'd neighbor features --------------------------
    for (int r = wave; r < KNBR; r += 8) {
        int idx = neighbor_list[n * KNBR + r];
        const _Float16* srow = nh16 + (size_t)idx * HDIM;
        for (int c = lane; c < HDIM; c += 32) sNBR[r * HDIM + c] = srow[c];
    }
    // ---- Stage 3: edge_attr_raw -> f16 -----------------------------------
    {
        const float* src = edge_attr_raw + (size_t)n * KNBR * DEDIM;
        for (int i = t; i < KNBR * DEDIM; i += 256) sEAin[i] = (_Float16)src[i];
    }
    __syncthreads();

    // ---- Stage 4: edge_attr = gelu(EAin @ W_ea + b) ----------------------
    for (int tile = wave; tile < 16; tile += 8) {
        int n0 = tile * 16;
        v8f c0 = {}, c1 = {};
        gemm_block2(c0, c1, sEAin, DEDIM, Wt_ea, DEDIM, n0, 0, DEDIM / 32, lane);
        int cn = n0 + (lane & 15), mb = ((lane >> 4) << 3);
        float bb = b_ea[cn];
#pragma unroll
        for (int r = 0; r < 8; r++) {
            sEA[(mb + r) * HDIM + cn]      = (_Float16)gelu_f(c0[r] + bb);
            sEA[(16 + mb + r) * HDIM + cn] = (_Float16)gelu_f(c1[r] + bb);
        }
    }
    // ---- Stage 5: node_hidden = gelu(NBR @ W_nh + b) ---------------------
    for (int tile = wave; tile < 16; tile += 8) {
        int n0 = tile * 16;
        v8f c0 = {}, c1 = {};
        gemm_block2(c0, c1, sNBR, HDIM, Wt_nh, HDIM, n0, 0, 8, lane);
        int cn = n0 + (lane & 15), mb = ((lane >> 4) << 3);
        float bb = b_nh[cn];
#pragma unroll
        for (int r = 0; r < 8; r++) {
            sNH[(mb + r) * HDIM + cn]      = (_Float16)gelu_f(c0[r] + bb);
            sNH[(16 + mb + r) * HDIM + cn] = (_Float16)gelu_f(c1[r] + bb);
        }
    }
    // ---- Stage 6: edge_hidden = gelu(EH @ W_eh + b) ----------------------
    for (int tile = wave; tile < 16; tile += 8) {
        int n0 = tile * 16;
        v8f c0 = {}, c1 = {};
        gemm_block2(c0, c1, sEH, HDIM, Wt_eh, HDIM, n0, 0, 8, lane);
        int cn = n0 + (lane & 15), mb = ((lane >> 4) << 3);
        float bb = b_eh[cn];
#pragma unroll
        for (int r = 0; r < 8; r++) {
            sEHID[(mb + r) * HDIM + cn]      = (_Float16)gelu_f(c0[r] + bb);
            sEHID[(16 + mb + r) * HDIM + cn] = (_Float16)gelu_f(c1[r] + bb);
        }
    }
    __syncthreads();

    // ---- Stage 7: gate/cand on m=[EA|NH], blend message ------------------
    for (int tile = wave; tile < 16; tile += 8) {
        int n0 = tile * 16;
        v8f cg0 = {}, cg1 = {}, cc0 = {}, cc1 = {};
        gemm_block2(cg0, cg1, sEA, HDIM, Wt_gate, H2, n0, 0,    8, lane);
        gemm_block2(cg0, cg1, sNH, HDIM, Wt_gate, H2, n0, HDIM, 8, lane);
        gemm_block2(cc0, cc1, sEA, HDIM, Wt_cand, H2, n0, 0,    8, lane);
        gemm_block2(cc0, cc1, sNH, HDIM, Wt_cand, H2, n0, HDIM, 8, lane);
        int cn = n0 + (lane & 15), mb = ((lane >> 4) << 3);
        float bg = b_gate[cn], bc = b_cand[cn];
#pragma unroll
        for (int r = 0; r < 8; r++) {
            float g  = sigmoid_f(cg0[r] + bg);
            float cd = tanh_fast(cc0[r] + bc);
            float eh = (float)sEHID[(mb + r) * HDIM + cn];
            sMSG[(mb + r) * HDIM + cn] = (_Float16)(g * cd + (1.0f - g) * eh);
            g  = sigmoid_f(cg1[r] + bg);
            cd = tanh_fast(cc1[r] + bc);
            eh = (float)sEHID[(16 + mb + r) * HDIM + cn];
            sMSG[(16 + mb + r) * HDIM + cn] = (_Float16)(g * cd + (1.0f - g) * eh);
        }
    }
    __syncthreads();

    // ---- Stage 8: qkv = MSG @ W_in + b (overlays Q/K/V on region 1) ------
    for (int tile = wave; tile < 48; tile += 8) {
        int nq0 = tile * 16;
        v8f c0 = {}, c1 = {};
        gemm_block2(c0, c1, sMSG, HDIM, Wt_in, HDIM, nq0, 0, 8, lane);
        int cn = nq0 + (lane & 15), mb = ((lane >> 4) << 3);
        float bb = b_in[cn];
        _Float16* dst; int cc;
        if (cn < HDIM)          { dst = sQ; cc = cn; }
        else if (cn < 2 * HDIM) { dst = sK; cc = cn - HDIM; }
        else                    { dst = sV; cc = cn - 2 * HDIM; }
#pragma unroll
        for (int r = 0; r < 8; r++) {
            dst[(mb + r) * HDIM + cc]      = (_Float16)(c0[r] + bb);
            dst[(16 + mb + r) * HDIM + cc] = (_Float16)(c1[r] + bb);
        }
    }
    __syncthreads();

    // ---- Stage 9: attention, one head per wave ---------------------------
    {
        const int h = wave;
        const float scal = attn_scalar[h];
        const float* ang = angle + ((size_t)n * HEADS + h) * KNBR * KNBR;
#pragma unroll
        for (int tn = 0; tn < 2; tn++) {
            v16h b  = load_b_frag(sK + h * DHEAD, HDIM, tn * 16, 0, lane);
            v16h a0 = load_a_frag(sQ + h * DHEAD, HDIM, 0,  0, lane);
            v16h a1 = load_a_frag(sQ + h * DHEAD, HDIM, 16, 0, lane);
            v8f c0 = {}, c1 = {};
            c0 = wmma_acc(c0, a0, b);
            c1 = wmma_acc(c1, a1, b);
            int cn = tn * 16 + (lane & 15), mb = ((lane >> 4) << 3);
            float am = (cn < cnt) ? 0.0f : BIG_NEG;
#pragma unroll
            for (int r = 0; r < 8; r++) {
                int cm = mb + r;
                sSC[(h * KNBR + cm) * KNBR + cn] =
                    c0[r] * 0.17677669529663687f + am + ang[cm * KNBR + cn] * scal;
                sSC[(h * KNBR + 16 + cm) * KNBR + cn] =
                    c1[r] * 0.17677669529663687f + am + ang[(16 + cm) * KNBR + cn] * scal;
            }
        }
        __syncthreads();
        {   // softmax: lane == query row (wave32)
            float* row = &sSC[(h * KNBR + lane) * KNBR];
            float mx = -3.4e38f;
#pragma unroll
            for (int j = 0; j < KNBR; j++) mx = fmaxf(mx, row[j]);
            float sum = 0.f;
#pragma unroll
            for (int j = 0; j < KNBR; j++) { float e = exp_fast(row[j] - mx); row[j] = e; sum += e; }
            float inv = rcp_fast(sum);
            _Float16* prow = &sP[(h * KNBR + lane) * KNBR];
#pragma unroll
            for (int j = 0; j < KNBR; j++) prow[j] = (_Float16)(row[j] * inv);
        }
        __syncthreads();
#pragma unroll
        for (int tn = 0; tn < 2; tn++) {
            v16h b  = load_b_frag_kmajor(sV + h * DHEAD, HDIM, tn * 16, 0, lane);
            v16h a0 = load_a_frag(sP + h * KNBR * KNBR, KNBR, 0,  0, lane);
            v16h a1 = load_a_frag(sP + h * KNBR * KNBR, KNBR, 16, 0, lane);
            v8f c0 = {}, c1 = {};
            c0 = wmma_acc(c0, a0, b);
            c1 = wmma_acc(c1, a1, b);
            int cn = tn * 16 + (lane & 15), mb = ((lane >> 4) << 3);
#pragma unroll
            for (int r = 0; r < 8; r++) {
                sCTX[(mb + r) * HDIM + h * DHEAD + cn]      = (_Float16)c0[r];
                sCTX[(16 + mb + r) * HDIM + h * DHEAD + cn] = (_Float16)c1[r];
            }
        }
    }
    __syncthreads();

    // ---- Stage 10: edge_output = CTX @ W_out + b (f32 in LDS) ------------
    for (int tile = wave; tile < 16; tile += 8) {
        int n0 = tile * 16;
        v8f c0 = {}, c1 = {};
        gemm_block2(c0, c1, sCTX, HDIM, Wt_out, HDIM, n0, 0, 8, lane);
        int cn = n0 + (lane & 15), mb = ((lane >> 4) << 3);
        float bb = b_out[cn];
#pragma unroll
        for (int r = 0; r < 8; r++) {
            sEO[(mb + r) * HDIM + cn]      = c0[r] + bb;
            sEO[(16 + mb + r) * HDIM + cn] = c1[r] + bb;
        }
    }
    __syncthreads();

    // ---- Stage 11: graph-attention aggregation to node ------------------
    {
        int kk = t >> 3, h = t & 7;
        float acc = 0.f;
#pragma unroll
        for (int d = 0; d < DHEAD; d++)
            acc += sEO[kk * HDIM + h * DHEAD + d] * attn_weight[h * DHEAD + d];
        acc = (acc > 0.f) ? acc : 0.2f * acc;                     // leaky relu
        sALPHA[kk * 8 + h] = (kk < cnt) ? acc : BIG_NEG;
    }
    __syncthreads();
    if (t < 8) {
        int h = t;
        float mx = -3.4e38f;
        for (int kk = 0; kk < KNBR; kk++) mx = fmaxf(mx, sALPHA[kk * 8 + h]);
        float sum = 0.f;
        for (int kk = 0; kk < KNBR; kk++) {
            float e = exp_fast(sALPHA[kk * 8 + h] - mx);
            sALPHA[kk * 8 + h] = e; sum += e;
        }
        float inv = rcp_fast(sum);
        for (int kk = 0; kk < KNBR; kk++) sALPHA[kk * 8 + h] *= inv;
    }
    __syncthreads();
    {
        int c = t, h = c >> 5;
        float acc = 0.f;
#pragma unroll
        for (int kk = 0; kk < KNBR; kk++) acc += sALPHA[kk * 8 + h] * sEO[kk * HDIM + c];
        node_resid[(size_t)n * HDIM + c] = acc + node_features[(size_t)n * HDIM + c];
    }
    __syncthreads();
    // ---- Stage 12: edge residual: sEO += edge_features -------------------
    {
        const float* ef = edge_features + (size_t)n * KNBR * HDIM;
        for (int i = t; i < KNBR * HDIM; i += 256) sEO[i] += ef[i];
    }
    __syncthreads();

    // ---- Stage 13: LN2e(edge_f) -> sE2 -----------------------------------
    {
        int r = t >> 3, q = t & 7;
        float xv[32], s0 = 0.f, s1 = 0.f;
#pragma unroll
        for (int j = 0; j < 32; j++) {
            float x = sEO[r * HDIM + q * 32 + j];
            xv[j] = x; s0 += x; s1 += x * x;
        }
        sRED[(r * 8 + q) * 2 + 0] = s0; sRED[(r * 8 + q) * 2 + 1] = s1;
        __syncthreads();
        float mu = 0.f, m2 = 0.f;
#pragma unroll
        for (int j = 0; j < 8; j++) { mu += sRED[(r * 8 + j) * 2]; m2 += sRED[(r * 8 + j) * 2 + 1]; }
        mu *= (1.0f / HDIM); m2 = m2 * (1.0f / HDIM) - mu * mu;
        float rs = rsqrtf(m2 + 1e-5f);
#pragma unroll
        for (int j = 0; j < 32; j++) {
            int c = q * 32 + j;
            sE2[r * HDIM + c] = (_Float16)((xv[j] - mu) * rs * ln2e_g[c] + ln2e_b[c]);
        }
    }
    __syncthreads();

    // ---- Stage 14: edge FFN + residual -> out ----------------------------
    for (int tile = wave; tile < 32; tile += 8) {                 // [32x256]@[256x512]
        int n0 = tile * 16;
        v8f c0 = {}, c1 = {};
        gemm_block2(c0, c1, sE2, HDIM, Wt_e1, HDIM, n0, 0, 8, lane);
        int cn = n0 + (lane & 15), mb = ((lane >> 4) << 3);
        float bb = be1[cn];
#pragma unroll
        for (int r = 0; r < 8; r++) {
            sT1[(mb + r) * H2 + cn]      = (_Float16)gelu_f(c0[r] + bb);
            sT1[(16 + mb + r) * H2 + cn] = (_Float16)gelu_f(c1[r] + bb);
        }
    }
    __syncthreads();
    for (int tile = wave; tile < 16; tile += 8) {                 // [32x512]@[512x256]
        int n0 = tile * 16;
        v8f c0 = {}, c1 = {};
        gemm_block2(c0, c1, sT1, H2, Wt_e2, H2, n0, 0, 16, lane);
        int cn = n0 + (lane & 15), mb = ((lane >> 4) << 3);
        float bb = be2[cn];
#pragma unroll
        for (int r = 0; r < 8; r++) {
            int cm = mb + r;
            out_edge[((size_t)n * KNBR + cm) * HDIM + cn]      = sEO[cm * HDIM + cn] + c0[r] + bb;
            out_edge[((size_t)n * KNBR + 16 + cm) * HDIM + cn] = sEO[(16 + cm) * HDIM + cn] + c1[r] + bb;
        }
    }
}

// ---------------------------------------------------------------------------
// Node FFN: 16 nodes per block. out = resid + FFN(LN(resid))
__global__ __launch_bounds__(256)
void node_ffn_kernel(const float* __restrict__ resid,
                     const float* __restrict__ g, const float* __restrict__ b,
                     const _Float16* __restrict__ Wt_n1, const float* __restrict__ bn1,
                     const _Float16* __restrict__ Wt_n2, const float* __restrict__ bn2,
                     float* __restrict__ out_node) {
    __shared__ float    sRES[16 * HDIM];
    __shared__ _Float16 sA[16 * HDIM];
    __shared__ _Float16 sT1[16 * H2];
    __shared__ float    sRED[16 * 16 * 2];
    int nb = blockIdx.x * 16;
    int t = threadIdx.x, lane = t & 31, wave = t >> 5;
    {
        int r = t >> 4, q = t & 15;
        const float* src = resid + (size_t)(nb + r) * HDIM + q * 16;
        float xv[16], s0 = 0.f, s1 = 0.f;
#pragma unroll
        for (int j = 0; j < 16; j++) {
            float x = src[j]; xv[j] = x;
            sRES[r * HDIM + q * 16 + j] = x;
            s0 += x; s1 += x * x;
        }
        sRED[(r * 16 + q) * 2 + 0] = s0; sRED[(r * 16 + q) * 2 + 1] = s1;
        __syncthreads();
        float mu = 0.f, m2 = 0.f;
#pragma unroll
        for (int j = 0; j < 16; j++) { mu += sRED[(r * 16 + j) * 2]; m2 += sRED[(r * 16 + j) * 2 + 1]; }
        mu *= (1.0f / HDIM); m2 = m2 * (1.0f / HDIM) - mu * mu;
        float rs = rsqrtf(m2 + 1e-5f);
#pragma unroll
        for (int j = 0; j < 16; j++) {
            int c = q * 16 + j;
            sA[r * HDIM + c] = (_Float16)((xv[j] - mu) * rs * g[c] + b[c]);
        }
    }
    __syncthreads();
    for (int tile = wave; tile < 32; tile += 8) {                 // [16x256]@[256x512]
        int n0 = tile * 16;
        v8f c = {};
        c = gemm_block(c, sA, HDIM, Wt_n1, HDIM, 0, n0, 0, 8, lane);
        int cn = n0 + (lane & 15), mb = ((lane >> 4) << 3);
        float bb = bn1[cn];
#pragma unroll
        for (int r = 0; r < 8; r++) sT1[(mb + r) * H2 + cn] = (_Float16)gelu_f(c[r] + bb);
    }
    __syncthreads();
    for (int tile = wave; tile < 16; tile += 8) {                 // [16x512]@[512x256]
        int n0 = tile * 16;
        v8f c = {};
        c = gemm_block(c, sT1, H2, Wt_n2, H2, 0, n0, 0, 16, lane);
        int cn = n0 + (lane & 15), mb = ((lane >> 4) << 3);
        float bb = bn2[cn];
#pragma unroll
        for (int r = 0; r < 8; r++) {
            int cm = mb + r;
            out_node[(size_t)(nb + cm) * HDIM + cn] = sRES[cm * HDIM + cn] + c[r] + bb;
        }
    }
}

// ---------------------------------------------------------------------------
extern "C" void kernel_launch(void* const* d_in, const int* in_sizes, int n_in,
                              void* d_out, int out_size, void* d_ws, size_t ws_size,
                              hipStream_t stream) {
    (void)in_sizes; (void)n_in; (void)out_size; (void)ws_size;
    const float* node_features = (const float*)d_in[0];
    const float* edge_features = (const float*)d_in[1];
    const float* edge_attr_raw = (const float*)d_in[2];
    const float* angle         = (const float*)d_in[3];
    const float* ln1n_g = (const float*)d_in[4];
    const float* ln1n_b = (const float*)d_in[5];
    const float* ln1e_g = (const float*)d_in[6];
    const float* ln1e_b = (const float*)d_in[7];
    const float* W_ea = (const float*)d_in[8];   const float* b_ea = (const float*)d_in[9];
    const float* W_nh = (const float*)d_in[10];  const float* b_nh = (const float*)d_in[11];
    const float* W_eh = (const float*)d_in[12];  const float* b_eh = (const float*)d_in[13];
    const float* W_gate = (const float*)d_in[14]; const float* b_gate = (const float*)d_in[15];
    const float* W_cand = (const float*)d_in[16]; const float* b_cand = (const float*)d_in[17];
    const float* attn_scalar = (const float*)d_in[18];
    const float* W_in = (const float*)d_in[19];  const float* b_in = (const float*)d_in[20];
    const float* W_out = (const float*)d_in[21]; const float* b_out = (const float*)d_in[22];
    const float* attn_weight = (const float*)d_in[23];
    const float* ln2n_g = (const float*)d_in[24]; const float* ln2n_b = (const float*)d_in[25];
    const float* ln2e_g = (const float*)d_in[26]; const float* ln2e_b = (const float*)d_in[27];
    const float* Wn1 = (const float*)d_in[28];   const float* bn1 = (const float*)d_in[29];
    const float* Wn2 = (const float*)d_in[30];   const float* bn2 = (const float*)d_in[31];
    const float* We1 = (const float*)d_in[32];   const float* be1 = (const float*)d_in[33];
    const float* We2 = (const float*)d_in[34];   const float* be2 = (const float*)d_in[35];
    const int* neighbor_list   = (const int*)d_in[36];
    const int* neighbor_counts = (const int*)d_in[37];

    // workspace carve-out
    char* ws = (char*)d_ws;
    size_t off = 0;
    auto alloc16 = [&](size_t halfs) -> _Float16* {
        _Float16* p = (_Float16*)(ws + off);
        off = (off + halfs * sizeof(_Float16) + 255) & ~(size_t)255;
        return p;
    };
    _Float16* Wt_ea   = alloc16((size_t)HDIM * DEDIM);
    _Float16* Wt_nh   = alloc16((size_t)HDIM * HDIM);
    _Float16* Wt_eh   = alloc16((size_t)HDIM * HDIM);
    _Float16* Wt_gate = alloc16((size_t)HDIM * H2);
    _Float16* Wt_cand = alloc16((size_t)HDIM * H2);
    _Float16* Wt_in   = alloc16((size_t)H3 * HDIM);
    _Float16* Wt_out  = alloc16((size_t)HDIM * HDIM);
    _Float16* Wt_n1   = alloc16((size_t)H2 * HDIM);
    _Float16* Wt_n2   = alloc16((size_t)HDIM * H2);
    _Float16* Wt_e1   = alloc16((size_t)H2 * HDIM);
    _Float16* Wt_e2   = alloc16((size_t)HDIM * H2);
    _Float16* nh16    = alloc16((size_t)NNODE * HDIM);
    float* node_resid = (float*)(ws + off);
    off += (size_t)NNODE * HDIM * sizeof(float);

    auto cvt = [&](const float* W, _Float16* Wt, int Kd, int Nd) {
        int tot = Kd * Nd;
        wt_convert_kernel<<<(tot + 255) / 256, 256, 0, stream>>>(W, Wt, Kd, Nd);
    };
    cvt(W_ea,   Wt_ea,   DEDIM, HDIM);
    cvt(W_nh,   Wt_nh,   HDIM,  HDIM);
    cvt(W_eh,   Wt_eh,   HDIM,  HDIM);
    cvt(W_gate, Wt_gate, H2,    HDIM);
    cvt(W_cand, Wt_cand, H2,    HDIM);
    cvt(W_in,   Wt_in,   HDIM,  H3);
    cvt(W_out,  Wt_out,  HDIM,  HDIM);
    cvt(Wn1,    Wt_n1,   HDIM,  H2);
    cvt(Wn2,    Wt_n2,   H2,    HDIM);
    cvt(We1,    Wt_e1,   HDIM,  H2);
    cvt(We2,    Wt_e2,   H2,    HDIM);

    node_ln_kernel<<<NNODE, 256, 0, stream>>>(node_features, ln1n_g, ln1n_b, nh16);

    const size_t dyn_lds = 219 * 1024;   // fits CDNA5's 320KB-per-workgroup LDS
    (void)hipFuncSetAttribute(reinterpret_cast<const void*>(edge_block_kernel),
                              hipFuncAttributeMaxDynamicSharedMemorySize, (int)dyn_lds);
    edge_block_kernel<<<NNODE, 256, dyn_lds, stream>>>(
        node_features, edge_features, edge_attr_raw, angle,
        ln1e_g, ln1e_b, b_ea, b_nh, b_eh, b_gate, b_cand, attn_scalar,
        b_in, b_out, attn_weight, ln2e_g, ln2e_b, be1, be2,
        Wt_ea, Wt_nh, Wt_eh, Wt_gate, Wt_cand, Wt_in, Wt_out, Wt_e1, Wt_e2,
        nh16, neighbor_list, neighbor_counts,
        node_resid, (float*)d_out + (size_t)NNODE * HDIM);

    float* out_node = (float*)d_out;
    node_ffn_kernel<<<NNODE / 16, 256, 0, stream>>>(
        node_resid, ln2n_g, ln2n_b, Wt_n1, bn1, Wt_n2, bn2, out_node);
}
